// HardAlignmentModule_4063039062716
// MI455X (gfx1250) — matrix operand order repaired and checked by reading, compiled, not run on hardware
//
#include <hip/hip_runtime.h>
#include <hip/hip_bf16.h>
#include <cstdint>
#include <cstddef>

// Problem constants from the reference (fixed shapes).
#define BATCH 4
#define C_IN  256
#define C_OUT 128
#define HW    4096

// Sim-kernel key-tile staging: 64 key rows per TDM tile, double buffered.
#define TN        64
#define NTILES    (HW / TN)             // 64
// TDM pads 4 DWORDs (16B) after every 64 DWORDs (= one 256B row):
// LDS row stride = 272B = 136 halfs -> 4-bank skew per row, conflict-free b128s.
#define LDS_ROW_H 136
#define TILE_H    (TN * LDS_ROW_H)      // halfs per buffer (8704)

typedef _Float16 half8  __attribute__((ext_vector_type(8)));
typedef _Float16 half16 __attribute__((ext_vector_type(16)));
typedef float    float8 __attribute__((ext_vector_type(8)));
typedef unsigned int uint32x4 __attribute__((ext_vector_type(4)));
typedef int          int32x4  __attribute__((ext_vector_type(4)));
typedef int          int32x8  __attribute__((ext_vector_type(8)));

union H16 { half16 v; half8 h[2]; };

// ---------------------------------------------------------------------------
// Kernel 0: transpose + convert  F[b][C_IN][HW] f32  ->  FT[b][HW][C_IN] f16
// grid: (HW/32, C_IN/32, BATCH*2)  block: (32, 8)
// ---------------------------------------------------------------------------
__global__ void k_transpose_cvt(const float* __restrict__ Fq,
                                const float* __restrict__ Fp,
                                _Float16* __restrict__ FqT,
                                _Float16* __restrict__ FpT) {
    __shared__ float s[32][33];
    const int t = blockIdx.z & 1;
    const int b = blockIdx.z >> 1;
    const float*    in  = (t ? Fp : Fq) + (size_t)b * C_IN * HW;
    _Float16*       out = (t ? FpT : FqT) + (size_t)b * HW * C_IN;
    const int n0 = blockIdx.x * 32;
    const int c0 = blockIdx.y * 32;
    const int tx = threadIdx.x, ty = threadIdx.y;
#pragma unroll
    for (int i = 0; i < 4; ++i)
        s[ty + 8 * i][tx] = in[(size_t)(c0 + ty + 8 * i) * HW + n0 + tx];
    __syncthreads();
#pragma unroll
    for (int i = 0; i < 4; ++i)
        out[(size_t)(n0 + ty + 8 * i) * C_IN + c0 + tx] =
            (_Float16)s[tx][ty + 8 * i];
}

// ---------------------------------------------------------------------------
// Kernel 1: convert W [C_OUT][C_IN] f32 -> f16.
// ---------------------------------------------------------------------------
__global__ void k_convert_w(const float* __restrict__ W,
                            _Float16* __restrict__ Wh) {
    int i = blockIdx.x * blockDim.x + threadIdx.x;
    if (i < C_OUT * C_IN) Wh[i] = (_Float16)W[i];
}

// ---------------------------------------------------------------------------
// Kernel 2: WMMA projection (q = W*x + bias) + per-pixel L2 normalize.
// One wave owns 16 pixels x full 128-channel column:
//   8 M-tiles x 8 K-chunks of v_wmma_f32_16x16x32_f16.
// grid: (HW/(16*8), 2, BATCH)  block: 256 (8 waves)
// ---------------------------------------------------------------------------
__global__ void k_proj_norm(const _Float16* __restrict__ FqT,
                            const _Float16* __restrict__ FpT,
                            const _Float16* __restrict__ Wh,
                            const float* __restrict__ bias,
                            _Float16* __restrict__ qh,
                            _Float16* __restrict__ ph) {
    const int wave = threadIdx.x >> 5;
    const int lane = threadIdx.x & 31;
    const int lm = lane & 15;
    const int lg = lane >> 4;
    const int t = blockIdx.y;
    const int b = blockIdx.z;
    const int n0 = (blockIdx.x * 8 + wave) * 16;

    const _Float16* ft  = (t ? FpT : FqT) + (size_t)b * HW * C_IN;
    _Float16*       out = (t ? ph : qh)   + (size_t)b * HW * C_OUT;

    const float8 zero = {0.f, 0.f, 0.f, 0.f, 0.f, 0.f, 0.f, 0.f};
    float8 c[8];
#pragma unroll
    for (int m = 0; m < 8; ++m) c[m] = zero;

    const _Float16* frow = ft + (size_t)(n0 + lm) * C_IN;
#pragma unroll
    for (int kc = 0; kc < 8; ++kc) {
        H16 Bv;
        const _Float16* bb = frow + kc * 32 + lg * 8;
        Bv.h[0] = *(const half8*)(bb);
        Bv.h[1] = *(const half8*)(bb + 16);
#pragma unroll
        for (int m = 0; m < 8; ++m) {
            H16 Av;
            const _Float16* aa =
                Wh + (size_t)(m * 16 + lm) * C_IN + kc * 32 + lg * 8;
            Av.h[0] = *(const half8*)(aa);
            Av.h[1] = *(const half8*)(aa + 16);
            c[m] = __builtin_amdgcn_wmma_f32_16x16x32_f16(
                false, Av.v, false, Bv.v, (short)0, c[m], false, false);
        }
    }

    float ss = 0.f;
#pragma unroll
    for (int m = 0; m < 8; ++m) {
#pragma unroll
        for (int r = 0; r < 8; ++r) {
            const int o = m * 16 + lg * 8 + r;
            float v = c[m][r] + bias[o];
            c[m][r] = v;
            ss += v * v;
        }
    }
    ss += __shfl_xor(ss, 16, 32);   // lane pair (l, l^16) covers all 128 rows
    const float inv = rsqrtf(ss);

#pragma unroll
    for (int m = 0; m < 8; ++m) {
        half8 hv;
#pragma unroll
        for (int r = 0; r < 8; ++r) hv[r] = (_Float16)(c[m][r] * inv);
        *(half8*)(out + (size_t)(n0 + lm) * C_OUT + m * 16 + lg * 8) = hv;
    }
}

// ---------------------------------------------------------------------------
// TDM helper: issue tensor_load_to_lds of a TN x C_OUT f16 tile (row-major,
// row stride C_OUT elements in memory) into LDS at lds_off, with 16B padding
// appended per 256B row (D# pad_interval=64 DW, pad_amount=4 DW).
// D# layout per CDNA5 ISA ch.8. This toolchain uses the 6-arg builtin form:
//   (uint32x4 g0, int32x8 g1, int32x4 g2, int32x4 g3, int32x8 extra, i32 cpol)
// ---------------------------------------------------------------------------
__device__ __forceinline__ void tdm_load_tile(const _Float16* gsrc,
                                              unsigned int lds_off) {
    const unsigned long long ga = (unsigned long long)(uintptr_t)gsrc;
    uint32x4 g0;
    g0[0] = 1u;                                    // count=1, user mode
    g0[1] = lds_off;                               // lds_addr
    g0[2] = (unsigned int)ga;                      // global_addr[31:0]
    g0[3] = (unsigned int)((ga >> 32) & 0x1FFFFFFu) | (2u << 30); // +type=2
    int32x8 g1;
    // data_size=1(2B) | pad_enable | pad_interval=5(64DW) | pad_amount=3(4DW)
    g1[0] = (int)((1u << 16) | (1u << 20) | (5u << 22) | (3u << 25));
    g1[1] = (int)((unsigned)C_OUT << 16);          // tensor_dim0 = 128
    g1[2] = (int)((unsigned)HW << 16);             // tensor_dim1 = 4096
    g1[3] = (int)((unsigned)C_OUT << 16);          // tile_dim0 = 128
    g1[4] = TN;                                    // tile_dim1 = 64
    g1[5] = C_OUT;                                 // tensor_dim0_stride = 128
    g1[6] = 0;
    g1[7] = 0;
    const int32x4 z4 = {0, 0, 0, 0};               // 2-D tensor: groups 2/3 off
    const int32x8 z8 = {0, 0, 0, 0, 0, 0, 0, 0};
    __builtin_amdgcn_tensor_load_to_lds(g0, g1, z4, z4, z8, 0);
}

// ---------------------------------------------------------------------------
// Kernel 3: fused sim-GEMM + row argmax, TDM double-buffered key tiles in LDS.
// Block = 4 waves; each wave owns 16 query rows; key tiles (64 rows) are
// DMA'd into LDS once per block and consumed by all 4 waves via ds_load_b128.
// grid: (HW/(16*4), BATCH)  block: 128
// ---------------------------------------------------------------------------
__global__ void k_sim_argmax(const _Float16* __restrict__ qh,
                             const _Float16* __restrict__ ph,
                             int* __restrict__ idx) {
    __shared__ __align__(128) _Float16 tile[2][TILE_H];

    const int wave = threadIdx.x >> 5;
    const int lane = threadIdx.x & 31;
    const int lm = lane & 15;
    const int lg = lane >> 4;
    const int b = blockIdx.y;
    const int m0 = (blockIdx.x * 4 + wave) * 16;

    const _Float16* q = qh + (size_t)b * HW * C_OUT;
    const _Float16* p = ph + (size_t)b * HW * C_OUT;

    const unsigned int lds0 = (unsigned int)(uintptr_t)(&tile[0][0]);
    const unsigned int lds1 = (unsigned int)(uintptr_t)(&tile[1][0]);

    // Kick off tile 0 DMA before touching anything else.
    if (wave == 0) tdm_load_tile(p, lds0);

    // Loop-invariant A fragments: query rows m0..m0+15, K = 0..127.
    H16 A[4];
#pragma unroll
    for (int kc = 0; kc < 4; ++kc) {
        const _Float16* aa = q + (size_t)(m0 + lm) * C_OUT + kc * 32 + lg * 8;
        A[kc].h[0] = *(const half8*)(aa);
        A[kc].h[1] = *(const half8*)(aa + 16);
    }

    float bestv[8];
    int   bestn[8];
#pragma unroll
    for (int r = 0; r < 8; ++r) { bestv[r] = -3.402823466e+38f; bestn[r] = 0; }

    for (int it = 0; it < NTILES; ++it) {
        if (wave == 0) __builtin_amdgcn_s_wait_tensorcnt(0);  // tile it landed
        __syncthreads();
        // Prefetch tile it+1 into the other buffer while computing tile it.
        // (End-of-iteration barrier below guarantees no reader still holds it.)
        if (wave == 0 && it + 1 < NTILES)
            tdm_load_tile(p + (size_t)(it + 1) * TN * C_OUT,
                          ((it + 1) & 1) ? lds1 : lds0);

        const _Float16* buf = &tile[it & 1][0];
        const int n0 = it * TN;
#pragma unroll
        for (int s = 0; s < TN / 16; ++s) {
            const _Float16* prow = buf + (size_t)(s * 16 + lm) * LDS_ROW_H;
            float8 c = {0.f, 0.f, 0.f, 0.f, 0.f, 0.f, 0.f, 0.f};
#pragma unroll
            for (int kc = 0; kc < 4; ++kc) {
                H16 Bv;                      // K = kc*32 + 8*lg + [0..7], +16
                const _Float16* bb = prow + kc * 32 + lg * 8;
                Bv.h[0] = *(const half8*)(bb);
                Bv.h[1] = *(const half8*)(bb + 16);
                c = __builtin_amdgcn_wmma_f32_16x16x32_f16(
                    false, A[kc].v, false, Bv.v, (short)0, c, false, false);
            }
            const int ncol = n0 + s * 16 + lm;
#pragma unroll
            for (int r = 0; r < 8; ++r) {
                float v = c[r];
                if (v > bestv[r]) { bestv[r] = v; bestn[r] = ncol; }
            }
        }
        __syncthreads();   // all reads of tile[it&1] done before it's rewritten
    }

    // Argmax across each 16-lane half (xor masks keep bit4 fixed).
#pragma unroll
    for (int off = 8; off >= 1; off >>= 1) {
#pragma unroll
        for (int r = 0; r < 8; ++r) {
            float ov = __shfl_xor(bestv[r], off, 32);
            int   on = __shfl_xor(bestn[r], off, 32);
            if (ov > bestv[r] || (ov == bestv[r] && on < bestn[r])) {
                bestv[r] = ov; bestn[r] = on;
            }
        }
    }
    if (lm == 0) {
#pragma unroll
        for (int r = 0; r < 8; ++r)
            idx[(size_t)b * HW + m0 + lg * 8 + r] = bestn[r];
    }
}

// ---------------------------------------------------------------------------
// Kernel 4: gather  out[b][ch][n] = Fp[b][ch][idx[b][n]]
// ---------------------------------------------------------------------------
__global__ void k_gather(const float* __restrict__ Fp,
                         const int* __restrict__ idx,
                         float* __restrict__ out) {
    size_t tid = (size_t)blockIdx.x * blockDim.x + threadIdx.x;
    const size_t total = (size_t)BATCH * C_IN * HW;
    if (tid >= total) return;
    const int n  = (int)(tid % HW);
    const int ch = (int)((tid / HW) % C_IN);
    const int b  = (int)(tid / ((size_t)HW * C_IN));
    const int j  = idx[(size_t)b * HW + n];
    out[tid] = Fp[((size_t)b * C_IN + ch) * HW + j];
}

// ---------------------------------------------------------------------------
extern "C" void kernel_launch(void* const* d_in, const int* in_sizes, int n_in,
                              void* d_out, int out_size, void* d_ws,
                              size_t ws_size, hipStream_t stream) {
    const float* Fq   = (const float*)d_in[0];
    const float* Fp   = (const float*)d_in[1];
    const float* W    = (const float*)d_in[2];
    const float* bias = (const float*)d_in[3];
    float* out = (float*)d_out;

    // Workspace layout (bytes): FqT 8MB | FpT 8MB | Wh 64KB | qh 4MB | ph 4MB
    //                           | idx 64KB   (total ~24.2 MB)
    char* ws = (char*)d_ws;
    _Float16* FqT = (_Float16*)(ws);
    _Float16* FpT = (_Float16*)(ws + (size_t)8 * 1024 * 1024);
    _Float16* Wh  = (_Float16*)(ws + (size_t)16 * 1024 * 1024);
    _Float16* qh  = (_Float16*)(ws + (size_t)16 * 1024 * 1024 + 64 * 1024);
    _Float16* ph  = (_Float16*)(ws + (size_t)20 * 1024 * 1024 + 64 * 1024);
    int*      idx = (int*)     (ws + (size_t)24 * 1024 * 1024 + 64 * 1024);

    k_transpose_cvt<<<dim3(HW / 32, C_IN / 32, BATCH * 2), dim3(32, 8, 1), 0,
                      stream>>>(Fq, Fp, FqT, FpT);
    k_convert_w<<<(C_OUT * C_IN + 255) / 256, 256, 0, stream>>>(W, Wh);
    k_proj_norm<<<dim3(HW / (16 * 8), 2, BATCH), 256, 0, stream>>>(
        FqT, FpT, Wh, bias, qh, ph);
    k_sim_argmax<<<dim3(HW / (16 * 4), BATCH, 1), 128, 0, stream>>>(qh, ph,
                                                                    idx);
    k_gather<<<(int)(((size_t)BATCH * C_IN * HW + 255) / 256), 256, 0,
               stream>>>(Fp, idx, out);
}
